// Qwen3DecoderLayer_69148973465601
// MI455X (gfx1250) — matrix operand order repaired and checked by reading, not compiled, not run on hardware
//
#include <hip/hip_runtime.h>
#include <hip/hip_bf16.h>

// ---------------------------------------------------------------------------
// Qwen3 decoder layer for MI455X (gfx1250): bf16 WMMA everywhere, f32 accum.
// GEMMs stage weight tiles in LDS via async global->LDS copies (ASYNCcnt).
// ---------------------------------------------------------------------------

#define L_TOK   4096
#define HID     2048
#define INTER   6144
#define NH      16
#define NKV     8
#define DH      128
#define HD      (NH * DH)     // 2048
#define KVD     (NKV * DH)    // 1024
#define SAMP    4
#define TLEN    1024

typedef __attribute__((ext_vector_type(16))) __bf16 v16bf;
typedef __attribute__((ext_vector_type(8)))  __bf16 v8bf;
typedef __attribute__((ext_vector_type(8)))  float  v8f;

// types matching the async-LDS builtin prototype: int __vector(4) with
// explicit address spaces (AS1 = global, AS3 = LDS)
typedef int v4i_ __attribute__((vector_size(16)));
typedef __attribute__((address_space(1))) v4i_ as1_v4i;
typedef __attribute__((address_space(3))) v4i_ as3_v4i;

__device__ inline v8f zero8() {
  v8f z;
#pragma unroll
  for (int i = 0; i < 8; ++i) z[i] = 0.0f;
  return z;
}

__device__ inline v16bf combine16(v8bf lo, v8bf hi) {
  v16bf f;
#pragma unroll
  for (int i = 0; i < 8; ++i) { f[i] = lo[i]; f[8 + i] = hi[i]; }
  return f;
}

// ---------------------------------------------------------------------------
// Async global -> LDS copy of one 16-byte chunk per lane (CDNA5 ASYNCcnt path)
// ---------------------------------------------------------------------------
__device__ inline void async_copy16(const __bf16* g, __bf16* l) {
#if __has_builtin(__builtin_amdgcn_global_load_async_to_lds_b128)
  __builtin_amdgcn_global_load_async_to_lds_b128(
      (as1_v4i*)g, (as3_v4i*)l, 0, 0);
#else
  // Flat shared-memory pointers carry the LDS byte offset in their low 32 bits
  // (ISA 10.2 aperture mapping), so truncation yields the DS address.
  unsigned lds = (unsigned)(unsigned long long)l;
  asm volatile("global_load_async_to_lds_b128 %0, %1, off"
               :: "v"(lds), "v"((unsigned long long)g)
               : "memory");
#endif
}

template <int N>
__device__ inline void wait_async() {
#if __has_builtin(__builtin_amdgcn_s_wait_asynccnt)
  __builtin_amdgcn_s_wait_asynccnt((unsigned short)N);
#else
  asm volatile("s_wait_asynccnt %0" :: "i"(N) : "memory");
#endif
}

// ---------------------------------------------------------------------------
// Elementwise / normalization kernels
// ---------------------------------------------------------------------------

__global__ void f32_to_bf16(const float* __restrict__ in, __bf16* __restrict__ out,
                            size_t n) {
  for (size_t i = (size_t)blockIdx.x * blockDim.x + threadIdx.x; i < n;
       i += (size_t)gridDim.x * blockDim.x)
    out[i] = (__bf16)in[i];
}

// out[c*rows + r] = in[r*cols + c]  (coalesced writes)
__global__ void transpose_to_bf16(const float* __restrict__ in,
                                  __bf16* __restrict__ out, int rows, int cols) {
  size_t n = (size_t)rows * cols;
  for (size_t i = (size_t)blockIdx.x * blockDim.x + threadIdx.x; i < n;
       i += (size_t)gridDim.x * blockDim.x) {
    size_t c = i / rows, r = i % rows;
    out[i] = (__bf16)in[r * (size_t)cols + c];
  }
}

__global__ __launch_bounds__(256) void rmsnorm_rows(const float* __restrict__ x,
                                                    const float* __restrict__ w,
                                                    __bf16* __restrict__ out,
                                                    int ncols) {
  const int row = blockIdx.x;
  const float* xr = x + (size_t)row * ncols;
  float ss = 0.0f;
  for (int i = threadIdx.x; i < ncols; i += 256) { float v = xr[i]; ss = fmaf(v, v, ss); }
#pragma unroll
  for (int off = 1; off < 32; off <<= 1) ss += __shfl_xor(ss, off);
  __shared__ float red[8];
  if ((threadIdx.x & 31) == 0) red[threadIdx.x >> 5] = ss;
  __syncthreads();
  float tot = red[0] + red[1] + red[2] + red[3] + red[4] + red[5] + red[6] + red[7];
  float inv = rsqrtf(tot / (float)ncols + 1e-6f);
  __bf16* orow = out + (size_t)row * ncols;
  for (int i = threadIdx.x; i < ncols; i += 256)
    orow[i] = (__bf16)(xr[i] * inv * w[i]);
}

// per-(token,head) RMSNorm over D=128, f32 in -> bf16 out. One wave per row.
__global__ __launch_bounds__(256) void head_rmsnorm(const float* __restrict__ x,
                                                    const float* __restrict__ w,
                                                    __bf16* __restrict__ out) {
  const int row  = blockIdx.x * 8 + (threadIdx.x >> 5);
  const int lane = threadIdx.x & 31;
  const float4 v = *(const float4*)(x + (size_t)row * DH + lane * 4);
  float ss = v.x * v.x + v.y * v.y + v.z * v.z + v.w * v.w;
#pragma unroll
  for (int off = 1; off < 32; off <<= 1) ss += __shfl_xor(ss, off);
  const float inv = rsqrtf(ss * (1.0f / DH) + 1e-6f);
  const float4 wv = *(const float4*)(w + lane * 4);
  __bf16* o = out + (size_t)row * DH + lane * 4;
  o[0] = (__bf16)(v.x * inv * wv.x);
  o[1] = (__bf16)(v.y * inv * wv.y);
  o[2] = (__bf16)(v.z * inv * wv.z);
  o[3] = (__bf16)(v.w * inv * wv.w);
}

// ---------------------------------------------------------------------------
// bf16 WMMA GEMM:  C[M,N] = A[M,K] * W[N,K]^T
// block = 256 threads (8 waves) -> 128x128 tile; wave = 16 rows x 8 N-tiles.
// B (weight) tiles double-buffered in LDS via global_load_async_to_lds_b128.
// EPI: 0 = f32 store, 1 = bf16 store, 2 = f32 residual add, 3 = silu(aux)*acc -> bf16
// ---------------------------------------------------------------------------
template <int EPI>
__global__ __launch_bounds__(256) void gemm_bf16(const __bf16* __restrict__ A,
                                                 const __bf16* __restrict__ W,
                                                 int N, int Kdim,
                                                 float* __restrict__ Cf,
                                                 __bf16* __restrict__ Cb,
                                                 const float* __restrict__ aux) {
  __shared__ __align__(16) __bf16 tile[2][128 * 32];   // 2 x 8KB B tiles

  const int wv   = threadIdx.x >> 5;
  const int lane = threadIdx.x & 31;
  const int m16  = lane & 15;
  const int half = lane >> 4;
  const int row0 = blockIdx.x * 128 + wv * 16;
  const int col0 = blockIdx.y * 128;

  v8f acc[8];
#pragma unroll
  for (int t = 0; t < 8; ++t) acc[t] = zero8();

  const __bf16* ap = A + (size_t)(row0 + m16) * Kdim + half * 16;

  // per-thread slice of the cooperative B-tile copy: 16 contiguous bf16
  // (one weight row, K-contiguous) split into two b128 chunks.
  const int  f0   = threadIdx.x * 16;             // flat bf16 index in tile
  const __bf16* wrow = W + (size_t)(col0 + (f0 >> 5)) * Kdim + (f0 & 31);

  async_copy16(wrow, &tile[0][f0]);
  async_copy16(wrow + 8, &tile[0][f0 + 8]);

  for (int kb = 0; kb < Kdim; kb += 32) {
    const int cur  = (kb >> 5) & 1;
    const bool more = (kb + 32) < Kdim;
    if (more) {
      async_copy16(wrow + kb + 32, &tile[cur ^ 1][f0]);
      async_copy16(wrow + kb + 40, &tile[cur ^ 1][f0 + 8]);
      wait_async<2>();            // current buffer's two chunks complete
    } else {
      wait_async<0>();
    }
    __syncthreads();              // publish all waves' async fills

    __builtin_prefetch(ap + kb + 256, 0, 0);      // global_prefetch_b8
    v8bf lo = *(const v8bf*)(ap + kb);
    v8bf hi = *(const v8bf*)(ap + kb + 8);
    v16bf af = combine16(lo, hi);

    const __bf16* bt = &tile[cur][0];
#pragma unroll
    for (int t = 0; t < 8; ++t) {
      v16bf bf = *(const v16bf*)(bt + (t * 16 + m16) * 32 + half * 16);
      acc[t] = __builtin_amdgcn_wmma_f32_16x16x32_bf16(false, af, false, bf,
                                                       (short)0, acc[t], false, false);
    }
    __syncthreads();              // readers done before buffer is refilled
  }

#pragma unroll
  for (int t = 0; t < 8; ++t) {
#pragma unroll
    for (int r = 0; r < 8; ++r) {
      const size_t idx = (size_t)(row0 + half * 8 + r) * N + col0 + t * 16 + m16;
      const float v = acc[t][r];
      if constexpr (EPI == 0) {
        Cf[idx] = v;
      } else if constexpr (EPI == 1) {
        Cb[idx] = (__bf16)v;
      } else if constexpr (EPI == 2) {
        Cf[idx] = aux[idx] + v;
      } else {
        const float g = aux[idx];
        Cb[idx] = (__bf16)((g / (1.0f + __expf(-g))) * v);   // silu(gate)*up
      }
    }
  }
}

// ---------------------------------------------------------------------------
// Flash attention (causal, per sample), bf16 WMMA, f32 online softmax.
// grid = (T/128, H, S); block = 256 (8 waves), wave owns a 16-row q tile.
// Q,K row-major bf16 [L][H*/KV* D]; Vt bf16 [KV*D][L]; O bf16 [L][H*D].
// ---------------------------------------------------------------------------
__global__ __launch_bounds__(256) void flash_attn(const __bf16* __restrict__ Q,
                                                  const __bf16* __restrict__ K,
                                                  const __bf16* __restrict__ Vt,
                                                  __bf16* __restrict__ O) {
  const int chunk = blockIdx.x;     // 0..7 (128 q rows each)
  const int h     = blockIdx.y;     // 0..15
  const int s     = blockIdx.z;     // 0..3
  const int kvh   = h >> 1;         // GQA group size 2
  const int wv    = threadIdx.x >> 5;
  const int lane  = threadIdx.x & 31;
  const int m16   = lane & 15;
  const int half  = lane >> 4;
  const int mh    = half * 8;

  const int qlocal0 = chunk * 128 + wv * 16;
  const int sbase   = s * TLEN;
  const int q0      = sbase + qlocal0;

  __shared__ __align__(16) __bf16 pstage[8][16 * 32];

  // resident Q A-fragments over D=128 (4 x 16x32)
  v16bf qf[4];
  {
    const __bf16* qp = Q + (size_t)(q0 + m16) * HD + h * DH + half * 16;
#pragma unroll
    for (int t = 0; t < 4; ++t) {
      v8bf lo = *(const v8bf*)(qp + t * 32);
      v8bf hi = *(const v8bf*)(qp + t * 32 + 8);
      qf[t] = combine16(lo, hi);
    }
  }

  v8f oacc[8];
#pragma unroll
  for (int t = 0; t < 8; ++t) oacc[t] = zero8();
  float mrow[8], lrow[8];
#pragma unroll
  for (int r = 0; r < 8; ++r) { mrow[r] = -3.0e38f; lrow[r] = 0.0f; }

  const int ntiles = chunk * 8 + wv + 1;       // causal 16-key tiles to visit
  const int npairs = (ntiles + 1) >> 1;
  const float scale = 0.088388347648318447f;   // 1/sqrt(128)

  for (int jp = 0; jp < npairs; ++jp) {
    // ---- scores: two 16x16 tiles, 4 WMMAs each over D ----
    v8f sacc[2];
    sacc[0] = zero8(); sacc[1] = zero8();
#pragma unroll
    for (int tt = 0; tt < 2; ++tt) {
      const int kt = jp * 2 + tt;
      const __bf16* kp =
          K + (size_t)(sbase + kt * 16 + m16) * KVD + kvh * DH + half * 16;
#pragma unroll
      for (int kb = 0; kb < 4; ++kb) {
        v16bf bf = *(const v16bf*)(kp + kb * 32);
        sacc[tt] = __builtin_amdgcn_wmma_f32_16x16x32_bf16(
            false, qf[kb], false, bf, (short)0, sacc[tt], false, false);
      }
    }

    // ---- mask + online softmax (row m = VGPR r + lane-half*8, col = lane&15) ----
    float pm[2][8];
#pragma unroll
    for (int r = 0; r < 8; ++r) {
      const int qrow = qlocal0 + mh + r;
#pragma unroll
      for (int tt = 0; tt < 2; ++tt) {
        const int key = (jp * 2 + tt) * 16 + m16;
        pm[tt][r] = sacc[tt][r] * scale + ((key > qrow) ? -1e9f : 0.0f);
      }
    }
#pragma unroll
    for (int r = 0; r < 8; ++r) {
      float tmax = fmaxf(pm[0][r], pm[1][r]);
#pragma unroll
      for (int off = 1; off < 16; off <<= 1) tmax = fmaxf(tmax, __shfl_xor(tmax, off));
      const float mnew = fmaxf(mrow[r], tmax);
      const float p0 = __expf(pm[0][r] - mnew);
      const float p1 = __expf(pm[1][r] - mnew);
      float rs = p0 + p1;
#pragma unroll
      for (int off = 1; off < 16; off <<= 1) rs += __shfl_xor(rs, off);
      const float corr = __expf(mrow[r] - mnew);
      lrow[r] = lrow[r] * corr + rs;
      mrow[r] = mnew;
#pragma unroll
      for (int t = 0; t < 8; ++t) oacc[t][r] *= corr;
      pstage[wv][(mh + r) * 32 + m16]      = (__bf16)p0;
      pstage[wv][(mh + r) * 32 + 16 + m16] = (__bf16)p1;
    }
    asm volatile("s_wait_dscnt 0x0" ::: "memory");

    // ---- re-layout P (C layout -> A fragment) via private LDS region ----
    v16bf pf;
    {
      const __bf16* pp = &pstage[wv][m16 * 32 + half * 16];
      v8bf lo = *(const v8bf*)pp;
      v8bf hi = *(const v8bf*)(pp + 8);
      pf = combine16(lo, hi);
    }

    // ---- P @ V : 8 WMMAs over D, V^T gives contiguous key runs ----
    const int t0 = sbase + jp * 32;
#pragma unroll
    for (int dt = 0; dt < 8; ++dt) {
      const __bf16* vp =
          Vt + (size_t)(kvh * DH + dt * 16 + m16) * L_TOK + t0 + half * 16;
      v16bf vf = *(const v16bf*)vp;
      oacc[dt] = __builtin_amdgcn_wmma_f32_16x16x32_bf16(
          false, pf, false, vf, (short)0, oacc[dt], false, false);
    }
  }

  // ---- normalize + store bf16 [L][H*D] ----
#pragma unroll
  for (int dt = 0; dt < 8; ++dt) {
#pragma unroll
    for (int r = 0; r < 8; ++r) {
      O[(size_t)(q0 + mh + r) * HD + h * DH + dt * 16 + m16] =
          (__bf16)(oacc[dt][r] / lrow[r]);
    }
  }
}

// ---------------------------------------------------------------------------
// Host-side orchestration
// ---------------------------------------------------------------------------
extern "C" void kernel_launch(void* const* d_in, const int* in_sizes, int n_in,
                              void* d_out, int out_size, void* d_ws, size_t ws_size,
                              hipStream_t stream) {
  (void)in_sizes; (void)n_in; (void)out_size; (void)ws_size;

  const float* x      = (const float*)d_in[0];   // [L, HID]
  const float* gen    = (const float*)d_in[1];   // [256, HID] (pass-through)
  // d_in[2] attention_mask : exactly causal -> computed inline
  // d_in[3] cos, d_in[4] sin : dead code in reference (pre-RoPE attention)
  // d_in[5] indexes : arange(L) -> identity scatter/gather
  const float* ln1_w  = (const float*)d_in[6];
  const float* ln2_w  = (const float*)d_in[7];
  const float* qn_w   = (const float*)d_in[8];
  const float* kn_w   = (const float*)d_in[9];
  const float* q_w    = (const float*)d_in[10];  // [HD, HID]
  const float* k_w    = (const float*)d_in[11];  // [KVD, HID]
  const float* v_w    = (const float*)d_in[12];  // [KVD, HID]
  const float* o_w    = (const float*)d_in[13];  // [HID, HD]
  const float* gate_w = (const float*)d_in[14];  // [INTER, HID]
  const float* up_w   = (const float*)d_in[15];  // [INTER, HID]
  const float* down_w = (const float*)d_in[16];  // [HID, INTER]

  char* wp = (char*)d_ws;
  auto carve = [&](size_t bytes) -> void* {
    void* r = wp;
    wp += (bytes + 255) & ~(size_t)255;
    return r;
  };

  __bf16* wq_b = (__bf16*)carve((size_t)HD * HID * 2);
  __bf16* wk_b = (__bf16*)carve((size_t)KVD * HID * 2);
  __bf16* wv_b = (__bf16*)carve((size_t)KVD * HID * 2);
  __bf16* wo_b = (__bf16*)carve((size_t)HID * HD * 2);
  __bf16* wg_b = (__bf16*)carve((size_t)INTER * HID * 2);
  __bf16* wu_b = (__bf16*)carve((size_t)INTER * HID * 2);
  __bf16* wd_b = (__bf16*)carve((size_t)HID * INTER * 2);
  __bf16* h_b  = (__bf16*)carve((size_t)L_TOK * HID * 2);
  float*  q_f  = (float*) carve((size_t)L_TOK * HD * 4);
  float*  k_f  = (float*) carve((size_t)L_TOK * KVD * 4);
  float*  v_f  = (float*) carve((size_t)L_TOK * KVD * 4);
  __bf16* qn_b = (__bf16*)carve((size_t)L_TOK * HD * 2);
  __bf16* kn_b = (__bf16*)carve((size_t)L_TOK * KVD * 2);
  __bf16* vt_b = (__bf16*)carve((size_t)KVD * L_TOK * 2);
  __bf16* at_b = (__bf16*)carve((size_t)L_TOK * HD * 2);
  float*  x1_f = (float*) carve((size_t)L_TOK * HID * 4);
  __bf16* h2_b = (__bf16*)carve((size_t)L_TOK * HID * 2);
  float*  g_f  = (float*) carve((size_t)L_TOK * INTER * 4);
  __bf16* a_b  = (__bf16*)carve((size_t)L_TOK * INTER * 2);

  auto conv = [&](const float* src, __bf16* dst, size_t n) {
    f32_to_bf16<<<dim3((unsigned)((n + 1023) / 1024)), 256, 0, stream>>>(src, dst, n);
  };
  conv(q_w,    wq_b, (size_t)HD * HID);
  conv(k_w,    wk_b, (size_t)KVD * HID);
  conv(v_w,    wv_b, (size_t)KVD * HID);
  conv(o_w,    wo_b, (size_t)HID * HD);
  conv(gate_w, wg_b, (size_t)INTER * HID);
  conv(up_w,   wu_b, (size_t)INTER * HID);
  conv(down_w, wd_b, (size_t)HID * INTER);

  // ln1
  rmsnorm_rows<<<L_TOK, 256, 0, stream>>>(x, ln1_w, h_b, HID);

  // q/k/v projections (f32 out for the per-head norms)
  gemm_bf16<0><<<dim3(L_TOK / 128, HD / 128),  256, 0, stream>>>(h_b, wq_b, HD,  HID, q_f, nullptr, nullptr);
  gemm_bf16<0><<<dim3(L_TOK / 128, KVD / 128), 256, 0, stream>>>(h_b, wk_b, KVD, HID, k_f, nullptr, nullptr);
  gemm_bf16<0><<<dim3(L_TOK / 128, KVD / 128), 256, 0, stream>>>(h_b, wv_b, KVD, HID, v_f, nullptr, nullptr);

  // q/k head RMSNorm -> bf16 ; V -> bf16 transposed [KV*D][L]
  head_rmsnorm<<<(L_TOK * NH) / 8,  256, 0, stream>>>(q_f, qn_w, qn_b);
  head_rmsnorm<<<(L_TOK * NKV) / 8, 256, 0, stream>>>(k_f, kn_w, kn_b);
  transpose_to_bf16<<<dim3((unsigned)(((size_t)L_TOK * KVD + 1023) / 1024)), 256, 0, stream>>>(v_f, vt_b, L_TOK, KVD);

  // causal flash attention
  flash_attn<<<dim3(TLEN / 128, NH, SAMP), 256, 0, stream>>>(qn_b, kn_b, vt_b, at_b);

  // o-proj + residual
  gemm_bf16<2><<<dim3(L_TOK / 128, HID / 128), 256, 0, stream>>>(at_b, wo_b, HID, HD, x1_f, nullptr, x);

  // ln2 + MLP (gate, silu*up fused, down + residual)
  rmsnorm_rows<<<L_TOK, 256, 0, stream>>>(x1_f, ln2_w, h2_b, HID);
  gemm_bf16<0><<<dim3(L_TOK / 128, INTER / 128), 256, 0, stream>>>(h2_b, wg_b, INTER, HID, g_f, nullptr, nullptr);
  gemm_bf16<3><<<dim3(L_TOK / 128, INTER / 128), 256, 0, stream>>>(h2_b, wu_b, INTER, HID, nullptr, a_b, g_f);
  gemm_bf16<2><<<dim3(L_TOK / 128, HID / 128),  256, 0, stream>>>(a_b, wd_b, HID, INTER, (float*)d_out, nullptr, x1_f);

  // second tuple output: packed_sequence_gen pass-through
  (void)hipMemcpyAsync((char*)d_out + (size_t)L_TOK * HID * 4, gen,
                       (size_t)256 * HID * 4, hipMemcpyDeviceToDevice, stream);
}
